// ParticleNet_90280212562690
// MI455X (gfx1250) — matrix operand order repaired and checked by reading, not compiled, not run on hardware
//
#include <hip/hip_runtime.h>
#include <stdint.h>

// ---------------------------------------------------------------------------
// ParticleNet forward for gfx1250 (CDNA5, wave32, WMMA f32_16x16x32_f16)
//
// Layouts:
//   features in (B,32,P) as given; all internal feature maps are POINT-MAJOR:
//   fts0  : (B*P, 32)
//   fused : (B*P, 448)   (block outputs concatenated at channel offsets 0/64/192)
//   edge-layer activations: row-major (B*P*16, co) f16 (pre-BN, stats read them)
// ---------------------------------------------------------------------------

typedef __attribute__((ext_vector_type(16))) _Float16 v16h;
typedef __attribute__((ext_vector_type(8)))  _Float16 v8h;
typedef __attribute__((ext_vector_type(8)))  float    v8f;

union HV { v16h v; v8h h[2]; };

#define AMODE_EDGE    0   // gather [x, nb-x] from point-major (B*P, ldc) at chOff
#define AMODE_BN_HALF 1   // relu(fma(half_src, scale[c], shift[c])), row-major (R,K)
#define AMODE_POINT   2   // point-major (B*P, ldc) at chOff, row r = b*128+p
#define AMODE_RAW_F32 3   // row-major f32 (R,K)
#define AMODE_RAW_H   4   // row-major f16 (R,K)

// ---------------------------------------------------------------------------
// WMMA GEMM: Out(M, Nreal) = A(M,K) * W(Nreal,K)^T.
// 256 threads = 8 waves; each wave owns a 16-row strip and ALL NT column tiles
// (A fragment built once per K-chunk, reused across NT WMMAs).
// grid.x = M/128.  K % 32 == 0.  NT*16 >= Nreal.
// ---------------------------------------------------------------------------
template<int AMODE, int NT, typename OT>
__global__ __launch_bounds__(256)
void wmma_gemm_kernel(const void* __restrict__ Asrc,
                      const float* __restrict__ W,      // (Nreal, K) f32 row-major
                      OT* __restrict__ Out, int ldOut,
                      const float* __restrict__ scaleA,
                      const float* __restrict__ shiftA,
                      const int*   __restrict__ nbr,    // (B*P, 16)
                      const float* __restrict__ bias,
                      int Nreal, int K,
                      int ldcA, int chOff, int Cfeat,
                      int epi)                           // 0 raw, 1 +bias, 2 +bias,relu
{
    __shared__ _Float16 BtT[NT * 16][40];                // [n][k], k contiguous, padded

    const int tid  = threadIdx.x;
    const int lane = tid & 31;
    const int wave = tid >> 5;
    const int rowBase = blockIdx.x * 128 + wave * 16;

    const int m  = lane & 15;                            // A row within tile
    const int kg = (lane >> 4) * 8;                      // A k-group (0 or 8)
    const int r  = rowBase + m;                          // this lane's A row

    const float*    Af = (const float*)Asrc;
    const _Float16* Ah = (const _Float16*)Asrc;

    long long baseX = 0, baseN = 0;
    if (AMODE == AMODE_EDGE) {
        int b  = r >> 11;                                // P*K = 2048
        int p  = (r >> 4) & 127;
        int kk = r & 15;
        int nb = nbr[((b << 7) + p) * 16 + kk];
        baseX = (long long)((b << 7) + p)  * ldcA + chOff;
        baseN = (long long)((b << 7) + nb) * ldcA + chOff;
    } else if (AMODE == AMODE_POINT) {
        baseX = (long long)r * ldcA + chOff;
    }

    const v8f vzero = {0.f, 0.f, 0.f, 0.f, 0.f, 0.f, 0.f, 0.f};
    v8f acc[NT];
#pragma unroll
    for (int t = 0; t < NT; ++t) acc[t] = vzero;

    for (int kc = 0; kc < K; kc += 32) {
        __syncthreads();                                  // BtT reuse fence
        // ---- stage weight K-chunk: 8 consecutive k per item ----
        // v8f (32B) global load -> cvt -> single 16B ds store. Branch-free.
#pragma unroll 2
        for (int i = tid; i < NT * 64; i += 256) {
            int n   = i >> 2;                             // output channel (row of W)
            int oct = (i & 3) * 8;                        // k octet within chunk
            int nsafe = (n < Nreal) ? n : (Nreal - 1);
            v8f wv = *(const v8f*)(W + (long long)nsafe * K + kc + oct);
            v8h hv;
#pragma unroll
            for (int j = 0; j < 8; ++j)
                hv[j] = (_Float16)((n < Nreal) ? wv[j] : 0.f);
            *(v8h*)(&BtT[n][oct]) = hv;
        }
        __syncthreads();

        // ---- A fragment (ISA 7.12.2 16-bit A layout: two 8-wide k groups) ----
        HV a;
        if (AMODE == AMODE_EDGE) {
            if (kc < Cfeat) {                             // uniform branch (Cfeat%32==0)
                v8f x0 = *(const v8f*)(Af + baseX + kc + kg);
                v8f x1 = *(const v8f*)(Af + baseX + kc + 16 + kg);
#pragma unroll
                for (int j = 0; j < 8; ++j) {
                    a.h[0][j] = (_Float16)x0[j];
                    a.h[1][j] = (_Float16)x1[j];
                }
            } else {
                int cc = kc - Cfeat;
                v8f n0 = *(const v8f*)(Af + baseN + cc + kg);
                v8f n1 = *(const v8f*)(Af + baseN + cc + 16 + kg);
                v8f x0 = *(const v8f*)(Af + baseX + cc + kg);
                v8f x1 = *(const v8f*)(Af + baseX + cc + 16 + kg);
#pragma unroll
                for (int j = 0; j < 8; ++j) {
                    a.h[0][j] = (_Float16)(n0[j] - x0[j]);
                    a.h[1][j] = (_Float16)(n1[j] - x1[j]);
                }
            }
        } else if (AMODE == AMODE_POINT) {
            v8f x0 = *(const v8f*)(Af + baseX + kc + kg);
            v8f x1 = *(const v8f*)(Af + baseX + kc + 16 + kg);
#pragma unroll
            for (int j = 0; j < 8; ++j) {
                a.h[0][j] = (_Float16)x0[j];
                a.h[1][j] = (_Float16)x1[j];
            }
        } else if (AMODE == AMODE_BN_HALF) {
            const _Float16* rowp = Ah + (long long)r * K;
            if (kc + 32 < K) __builtin_prefetch(rowp + kc + 32, 0, 3);
            v8h x0 = *(const v8h*)(rowp + kc + kg);
            v8h x1 = *(const v8h*)(rowp + kc + 16 + kg);
            v8f s0 = *(const v8f*)(scaleA + kc + kg);
            v8f s1 = *(const v8f*)(scaleA + kc + 16 + kg);
            v8f h0 = *(const v8f*)(shiftA + kc + kg);
            v8f h1 = *(const v8f*)(shiftA + kc + 16 + kg);
#pragma unroll
            for (int j = 0; j < 8; ++j) {
                a.h[0][j] = (_Float16)fmaxf(fmaf((float)x0[j], s0[j], h0[j]), 0.f);
                a.h[1][j] = (_Float16)fmaxf(fmaf((float)x1[j], s1[j], h1[j]), 0.f);
            }
        } else if (AMODE == AMODE_RAW_F32) {
            const float* rowp = Af + (long long)r * K;
            v8f x0 = *(const v8f*)(rowp + kc + kg);
            v8f x1 = *(const v8f*)(rowp + kc + 16 + kg);
#pragma unroll
            for (int j = 0; j < 8; ++j) {
                a.h[0][j] = (_Float16)x0[j];
                a.h[1][j] = (_Float16)x1[j];
            }
        } else {                                          // AMODE_RAW_H
            const _Float16* rowp = Ah + (long long)r * K;
            a.h[0] = *(const v8h*)(rowp + kc + kg);
            a.h[1] = *(const v8h*)(rowp + kc + 16 + kg);
        }

        // ---- NT WMMAs reusing the A fragment ----
        const int nn  = lane & 15;
        const int kgb = (lane >> 4) * 8;
#pragma unroll
        for (int nt = 0; nt < NT; ++nt) {
            HV bf;
            bf.h[0] = *(const v8h*)(&BtT[nt * 16 + nn][kgb]);
            bf.h[1] = *(const v8h*)(&BtT[nt * 16 + nn][kgb + 16]);
            acc[nt] = __builtin_amdgcn_wmma_f32_16x16x32_f16(
                          false, a.v, false, bf.v, (short)0, acc[nt], false, false);
        }
    }

    // ---- epilogue (C/D layout: lanes 0-15 M=rr, lanes 16-31 M=8+rr) ----
    const int col    = lane & 15;
    const int rowOff = (lane >> 4) * 8;
#pragma unroll
    for (int nt = 0; nt < NT; ++nt) {
        int n = nt * 16 + col;
        if (n < Nreal) {
#pragma unroll
            for (int rr = 0; rr < 8; ++rr) {
                int orow = rowBase + rowOff + rr;
                float v = acc[nt][rr];
                if (epi >= 1) v += bias[n];
                if (epi == 2) v = fmaxf(v, 0.f);
                Out[(long long)orow * ldOut + n] = (OT)v;
            }
        }
    }
}

// ---------------------------------------------------------------------------
// BN stats, deterministic 2-phase, coalesced. Phase 1: blockDim == C, each
// thread owns channel c, streams rows. Phase 2: fixed-order reduce of 256
// per-block partials -> scale = g*rstd, shift = b - mean*scale.
// ---------------------------------------------------------------------------
template<typename T>
__global__ void stats_partial_kernel(const T* __restrict__ X, long long R, int C,
                                     float* __restrict__ psum, float* __restrict__ psq)
{
    int c = threadIdx.x;                                  // blockDim.x == C
    long long rowsPer = R >> 8;                           // R / 256
    long long r0 = (long long)blockIdx.x * rowsPer;
    float sum = 0.f, sq = 0.f;
    for (long long j = 0; j < rowsPer; ++j) {
        float v = (float)X[(r0 + j) * C + c];
        sum += v; sq += v * v;
    }
    psum[(long long)blockIdx.x * C + c] = sum;
    psq [(long long)blockIdx.x * C + c] = sq;
}

__global__ __launch_bounds__(256)
void stats_finalize_kernel(const float* __restrict__ psum, const float* __restrict__ psq,
                           int C, float invR,
                           const float* __restrict__ gamma, const float* __restrict__ beta,
                           float* __restrict__ scale, float* __restrict__ shift)
{
    int c = blockIdx.x;
    int t = threadIdx.x;
    __shared__ float s1[256], s2[256];
    s1[t] = psum[(long long)t * C + c];
    s2[t] = psq [(long long)t * C + c];
    __syncthreads();
    for (int s = 128; s > 0; s >>= 1) {
        if (t < s) { s1[t] += s1[t + s]; s2[t] += s2[t + s]; }
        __syncthreads();
    }
    if (t == 0) {
        float mean = s1[0] * invR;
        float var  = s2[0] * invR - mean * mean;
        float rstd = rsqrtf(fmaxf(var, 0.f) + 1e-5f);
        float sc   = gamma[c] * rstd;
        scale[c] = sc;
        shift[c] = beta[c] - mean * sc;
    }
}

// mask / masked copies (point-major outputs) / coord_shift
__global__ void prep_kernel(const float* __restrict__ points,
                            const float* __restrict__ feat,
                            float* __restrict__ mpts, float* __restrict__ mfeat,
                            float* __restrict__ mask, float* __restrict__ cshift)
{
    int bp = blockIdx.x * 256 + threadIdx.x;
    if (bp >= 128 * 128) return;
    int b = bp >> 7, p = bp & 127;
    float s = 0.f;
    for (int c = 0; c < 32; ++c)
        s += fabsf(feat[((long long)(b * 32 + c)) * 128 + p]);
    float m = (s != 0.f) ? 1.f : 0.f;
    mask[bp]   = m;
    cshift[bp] = (m == 0.f) ? 1e9f : 0.f;
    for (int d = 0; d < 2; ++d)
        mpts[(long long)bp * 2 + d] =
            points[((long long)(b * 2 + d)) * 128 + p] * m;
    for (int c = 0; c < 32; ++c)
        mfeat[(long long)bp * 32 + c] =
            feat[((long long)(b * 32 + c)) * 128 + p] * m;
}

__global__ __launch_bounds__(128)
void counts_kernel(const float* __restrict__ mask, float* __restrict__ counts)
{
    __shared__ float sm[128];
    int b = blockIdx.x;
    sm[threadIdx.x] = mask[b * 128 + threadIdx.x];
    __syncthreads();
    for (int s = 64; s > 0; s >>= 1) {
        if ((int)threadIdx.x < s) sm[threadIdx.x] += sm[threadIdx.x + s];
        __syncthreads();
    }
    if (threadIdx.x == 0) counts[b] = fmaxf(sm[0], 1.f);
}

// fts0[bp*32+c] = BN(mfeat) * mask
__global__ void bnapply_kernel(const float* __restrict__ mfeat,
                               const float* __restrict__ scale,
                               const float* __restrict__ shift,
                               const float* __restrict__ mask,
                               float* __restrict__ fts0)
{
    long long i = (long long)blockIdx.x * 256 + threadIdx.x;
    if (i >= 128LL * 128 * 32) return;
    int c  = (int)(i & 31);
    int bp = (int)(i >> 5);
    fts0[i] = fmaf(mfeat[i], scale[c], shift[c]) * mask[bp];
}

// ---------------------------------------------------------------------------
// KNN on point-major coords (B*P, ldc) at chOff. One workgroup per (b,p),
// 128 threads (one per candidate q). 17 argmax rounds (lowest index on ties).
// ---------------------------------------------------------------------------
__global__ __launch_bounds__(128)
void knn_kernel(const float* __restrict__ X, int ldc, int chOff, int D,
                const float* __restrict__ cshift, int* __restrict__ idxOut)
{
    int bp = blockIdx.x;
    int b = bp >> 7;
    int q = threadIdx.x;
    __shared__ float xp[128];
    __shared__ float pd[128];
    __shared__ float rv[128];
    __shared__ int   ri[128];

    for (int d = q; d < D; d += 128)                      // coalesced (same row)
        xp[d] = X[(long long)bp * ldc + chOff + d];
    __syncthreads();

    float csp = cshift[bp];
    float csq = cshift[b * 128 + q];
    const float* xqrow = X + (long long)(b * 128 + q) * ldc + chOff;
    float acc = 0.f;
    for (int d = 0; d < D; ++d) {                         // per-lane row stream
        float diff = (xp[d] + csp) - (xqrow[d] + csq);
        acc = fmaf(diff, diff, acc);
    }
    pd[q] = -acc;
    __syncthreads();

    for (int t = 0; t <= 16; ++t) {
        rv[q] = pd[q]; ri[q] = q;
        __syncthreads();
        for (int s = 64; s > 0; s >>= 1) {
            if (q < s) {
                float v2 = rv[q + s]; int i2 = ri[q + s];
                if (v2 > rv[q] || (v2 == rv[q] && i2 < ri[q])) { rv[q] = v2; ri[q] = i2; }
            }
            __syncthreads();
        }
        int win = ri[0];
        if (t > 0 && q == 0) idxOut[(bp << 4) + (t - 1)] = win;
        if (q == win) pd[q] = -3.4e38f;
        __syncthreads();
    }
}

// relu(BN(sc) + mean_k(relu(BN(conv3)))) * mask -> fused (B*P, 448) at chOff
__global__ void combine_kernel(const _Float16* __restrict__ out3,
                               const float* __restrict__ s3, const float* __restrict__ sh3,
                               const float* __restrict__ scOut,
                               const float* __restrict__ ssc, const float* __restrict__ shsc,
                               const float* __restrict__ mask,
                               float* __restrict__ fused,
                               int co, int chOff, long long total)
{
    long long i = (long long)blockIdx.x * 256 + threadIdx.x;
    if (i >= total) return;
    int c = (int)(i % co);
    long long bp = i / co;
    long long base = (bp << 4) * co + c;
    float m = 0.f;
#pragma unroll
    for (int kk = 0; kk < 16; ++kk)
        m += fmaxf(fmaf((float)out3[base + (long long)kk * co], s3[c], sh3[c]), 0.f);
    m *= (1.f / 16.f);
    float sc = fmaf(scOut[bp * co + c], ssc[c], shsc[c]);
    float v  = fmaxf(sc + m, 0.f) * mask[bp];
    fused[bp * 448 + chOff + c] = v;                      // coalesced over c
}

// x[b][c] = sum_p relu(BN(fuse)) * mask / counts
__global__ void pool_kernel(const float* __restrict__ fo,
                            const float* __restrict__ sf, const float* __restrict__ shf,
                            const float* __restrict__ mask,
                            const float* __restrict__ counts,
                            float* __restrict__ x)
{
    int i = blockIdx.x * 256 + threadIdx.x;
    if (i >= 128 * 384) return;
    int b = i / 384, c = i % 384;
    float s = 0.f;
    for (int p = 0; p < 128; ++p) {
        int bp = b * 128 + p;
        float v = fmaxf(fmaf(fo[(long long)bp * 384 + c], sf[c], shf[c]), 0.f);
        s += v * mask[bp];
    }
    x[i] = s / counts[b];
}

// ---------------------------------------------------------------------------
extern "C" void kernel_launch(void* const* d_in, const int* in_sizes, int n_in,
                              void* d_out, int out_size, void* d_ws, size_t ws_size,
                              hipStream_t stream)
{
    (void)in_sizes; (void)n_in; (void)out_size; (void)ws_size;

    // ---- unpack inputs (setup_inputs dict insertion order) ----
    int ii = 0;
    const float* points   = (const float*)d_in[ii++];
    const float* features = (const float*)d_in[ii++];
    const float* bnf_g = (const float*)d_in[ii++];
    const float* bnf_b = (const float*)d_in[ii++];
    struct Blk {
        const float* w[3];
        const float* g[3]; const float* b[3];
        const float* scw; const float* scg; const float* scb;
    } blk[3];
    for (int i = 0; i < 3; ++i) {
        for (int j = 0; j < 3; ++j) blk[i].w[j] = (const float*)d_in[ii++];
        for (int j = 0; j < 3; ++j) {
            blk[i].g[j] = (const float*)d_in[ii++];
            blk[i].b[j] = (const float*)d_in[ii++];
        }
        blk[i].scw = (const float*)d_in[ii++];
        blk[i].scg = (const float*)d_in[ii++];
        blk[i].scb = (const float*)d_in[ii++];
    }
    const float* fuse_w = (const float*)d_in[ii++];
    const float* fuse_g = (const float*)d_in[ii++];
    const float* fuse_b = (const float*)d_in[ii++];
    const float* fc1_w  = (const float*)d_in[ii++];
    const float* fc1_b  = (const float*)d_in[ii++];
    const float* fc2_w  = (const float*)d_in[ii++];
    const float* fc2_b  = (const float*)d_in[ii++];

    // ---- workspace bump allocator (256B aligned) ----
    char* wsp = (char*)d_ws;
    size_t off = 0;
    auto take = [&](size_t bytes) -> void* {
        void* r = wsp + off;
        off += (bytes + 255) & ~(size_t)255;
        return r;
    };
    float*    mpts    = (float*)take(16384LL * 2 * 4);
    float*    mfeat   = (float*)take(16384LL * 32 * 4);
    float*    mask    = (float*)take(16384 * 4);
    float*    cshift  = (float*)take(16384 * 4);
    float*    counts  = (float*)take(128 * 4);
    float*    fts0    = (float*)take(16384LL * 32 * 4);
    float*    fused   = (float*)take(16384LL * 448 * 4);
    int*      nbr     = (int*)  take(16384LL * 16 * 4);
    _Float16* bufA    = (_Float16*)take(262144LL * 256 * 2);
    _Float16* bufB    = (_Float16*)take(262144LL * 256 * 2);
    float*    scOut   = (float*)take(16384LL * 256 * 4);
    float*    fuseOut = (float*)take(16384LL * 384 * 4);
    float*    xbuf    = (float*)take(128LL * 384 * 4);
    _Float16* hbuf    = (_Float16*)take(128LL * 256 * 2);
    float*    psum    = (float*)take(256LL * 384 * 4);
    float*    psq     = (float*)take(256LL * 384 * 4);
    float* s1  = (float*)take(512 * 4); float* sh1  = (float*)take(512 * 4);
    float* s2  = (float*)take(512 * 4); float* sh2  = (float*)take(512 * 4);
    float* s3  = (float*)take(512 * 4); float* sh3  = (float*)take(512 * 4);
    float* ssc = (float*)take(512 * 4); float* shsc = (float*)take(512 * 4);
    float* sf  = (float*)take(512 * 4); float* shf  = (float*)take(512 * 4);

    auto statsH = [&](const _Float16* X, long long R, int C,
                      const float* g, const float* b, float* sc, float* sh) {
        stats_partial_kernel<_Float16><<<256, C, 0, stream>>>(X, R, C, psum, psq);
        stats_finalize_kernel<<<C, 256, 0, stream>>>(psum, psq, C, 1.f / (float)R, g, b, sc, sh);
    };
    auto statsF = [&](const float* X, long long R, int C,
                      const float* g, const float* b, float* sc, float* sh) {
        stats_partial_kernel<float><<<256, C, 0, stream>>>(X, R, C, psum, psq);
        stats_finalize_kernel<<<C, 256, 0, stream>>>(psum, psq, C, 1.f / (float)R, g, b, sc, sh);
    };

    // ---- preprocessing ----
    prep_kernel<<<64, 256, 0, stream>>>(points, features, mpts, mfeat, mask, cshift);
    counts_kernel<<<128, 128, 0, stream>>>(mask, counts);
    statsF(mfeat, 16384LL, 32, bnf_g, bnf_b, s1, sh1);
    bnapply_kernel<<<2048, 256, 0, stream>>>(mfeat, s1, sh1, mask, fts0);

    const int Cin[3]   = {32, 64, 128};
    const int Co[3]    = {64, 128, 256};
    const int ChOut[3] = {0, 64, 192};
    const long long RE = 262144;                          // B*P*K rows

    for (int i = 0; i < 3; ++i) {
        const float* fin; int ldc, chIn;
        if (i == 0) { fin = fts0;  ldc = 32;  chIn = 0; }
        else        { fin = fused; ldc = 448; chIn = ChOut[i - 1]; }
        const float* pts; int pldc, pch, D;
        if (i == 0) { pts = mpts;  pldc = 2;   pch = 0;            D = 2; }
        else        { pts = fused; pldc = 448; pch = ChOut[i - 1]; D = Cin[i]; }
        knn_kernel<<<16384, 128, 0, stream>>>(pts, pldc, pch, D, cshift, nbr);

        int co = Co[i], cin2 = 2 * Cin[i];

        // layer 1: edge gather GEMM
        switch (co) {
        case 64:  wmma_gemm_kernel<AMODE_EDGE,  4, _Float16><<<2048, 256, 0, stream>>>(
                      fin, blk[i].w[0], bufA, co, nullptr, nullptr, nbr, nullptr,
                      co, cin2, ldc, chIn, Cin[i], 0); break;
        case 128: wmma_gemm_kernel<AMODE_EDGE,  8, _Float16><<<2048, 256, 0, stream>>>(
                      fin, blk[i].w[0], bufA, co, nullptr, nullptr, nbr, nullptr,
                      co, cin2, ldc, chIn, Cin[i], 0); break;
        default:  wmma_gemm_kernel<AMODE_EDGE, 16, _Float16><<<2048, 256, 0, stream>>>(
                      fin, blk[i].w[0], bufA, co, nullptr, nullptr, nbr, nullptr,
                      co, cin2, ldc, chIn, Cin[i], 0); break;
        }
        statsH(bufA, RE, co, blk[i].g[0], blk[i].b[0], s1, sh1);

        // layers 2,3: BN+ReLU fused into A operand
        switch (co) {
        case 64:
            wmma_gemm_kernel<AMODE_BN_HALF, 4, _Float16><<<2048, 256, 0, stream>>>(
                bufA, blk[i].w[1], bufB, co, s1, sh1, nullptr, nullptr, co, co, 0, 0, 0, 0);
            break;
        case 128:
            wmma_gemm_kernel<AMODE_BN_HALF, 8, _Float16><<<2048, 256, 0, stream>>>(
                bufA, blk[i].w[1], bufB, co, s1, sh1, nullptr, nullptr, co, co, 0, 0, 0, 0);
            break;
        default:
            wmma_gemm_kernel<AMODE_BN_HALF, 16, _Float16><<<2048, 256, 0, stream>>>(
                bufA, blk[i].w[1], bufB, co, s1, sh1, nullptr, nullptr, co, co, 0, 0, 0, 0);
            break;
        }
        statsH(bufB, RE, co, blk[i].g[1], blk[i].b[1], s2, sh2);

        switch (co) {
        case 64:
            wmma_gemm_kernel<AMODE_BN_HALF, 4, _Float16><<<2048, 256, 0, stream>>>(
                bufB, blk[i].w[2], bufA, co, s2, sh2, nullptr, nullptr, co, co, 0, 0, 0, 0);
            break;
        case 128:
            wmma_gemm_kernel<AMODE_BN_HALF, 8, _Float16><<<2048, 256, 0, stream>>>(
                bufB, blk[i].w[2], bufA, co, s2, sh2, nullptr, nullptr, co, co, 0, 0, 0, 0);
            break;
        default:
            wmma_gemm_kernel<AMODE_BN_HALF, 16, _Float16><<<2048, 256, 0, stream>>>(
                bufB, blk[i].w[2], bufA, co, s2, sh2, nullptr, nullptr, co, co, 0, 0, 0, 0);
            break;
        }
        statsH(bufA, RE, co, blk[i].g[2], blk[i].b[2], s3, sh3);

        // shortcut GEMM over points
        switch (co) {
        case 64:  wmma_gemm_kernel<AMODE_POINT,  4, float><<<128, 256, 0, stream>>>(
                      fin, blk[i].scw, scOut, co, nullptr, nullptr, nullptr, nullptr,
                      co, Cin[i], ldc, chIn, 0, 0); break;
        case 128: wmma_gemm_kernel<AMODE_POINT,  8, float><<<128, 256, 0, stream>>>(
                      fin, blk[i].scw, scOut, co, nullptr, nullptr, nullptr, nullptr,
                      co, Cin[i], ldc, chIn, 0, 0); break;
        default:  wmma_gemm_kernel<AMODE_POINT, 16, float><<<128, 256, 0, stream>>>(
                      fin, blk[i].scw, scOut, co, nullptr, nullptr, nullptr, nullptr,
                      co, Cin[i], ldc, chIn, 0, 0); break;
        }
        statsF(scOut, 16384LL, co, blk[i].scg, blk[i].scb, ssc, shsc);

        long long tot = 16384LL * co;
        combine_kernel<<<(unsigned)((tot + 255) / 256), 256, 0, stream>>>(
            bufA, s3, sh3, scOut, ssc, shsc, mask, fused, co, ChOut[i], tot);
    }

    // ---- fuse conv + BN + pool ----
    wmma_gemm_kernel<AMODE_POINT, 24, float><<<128, 256, 0, stream>>>(
        fused, fuse_w, fuseOut, 384, nullptr, nullptr, nullptr, nullptr,
        384, 448, 448, 0, 0, 0);
    statsF(fuseOut, 16384LL, 384, fuse_g, fuse_b, sf, shf);
    pool_kernel<<<(128 * 384 + 255) / 256, 256, 0, stream>>>(
        fuseOut, sf, shf, mask, counts, xbuf);

    // ---- FC head ----
    wmma_gemm_kernel<AMODE_RAW_F32, 16, _Float16><<<1, 256, 0, stream>>>(
        xbuf, fc1_w, hbuf, 256, nullptr, nullptr, nullptr, fc1_b,
        256, 384, 0, 0, 0, 2);
    wmma_gemm_kernel<AMODE_RAW_H, 1, float><<<1, 256, 0, stream>>>(
        hbuf, fc2_w, (float*)d_out, 10, nullptr, nullptr, nullptr, fc2_b,
        10, 256, 0, 0, 0, 1);
}
